// AdaptiveSparseMoE_72533407695264
// MI455X (gfx1250) — matrix operand order repaired and checked by reading, not compiled, not run on hardware
//
#include <hip/hip_runtime.h>
#include <hip/hip_bf16.h>

// ---------------------------------------------------------------------------
// AdaptiveSparseMoE for MI455X (gfx1250, wave32, WMMA).
//
// Shapes: B=4 S=2048 D=1024 E=4 F=4096, N = B*S = 8192 tokens.
// Compute-bound (550 GFLOP vs ~200MB HBM traffic; weights L2-resident) ->
// both GEMMs run through v_wmma_f32_16x16x32_f16 with fp32 accumulation.
// LDS kept to ~128.5KB so two 8-wave blocks fit per 320KB WGP (16 waves).
// Workspace layout (assumes ws_size >= ~85 MB):
//   [0)          Xh   : N*D f16            (16 MB)
//   [16MB)       W1t  : E*F*D f16 (W1^T)   (32 MB)
//   [48MB)       W2t  : E*D*F f16 (W2^T)   (32 MB)
//   [80MB)       mask : N*E f32 binary     (128 KB)
// ---------------------------------------------------------------------------

typedef __attribute__((ext_vector_type(16))) _Float16 v16h;
typedef __attribute__((ext_vector_type(8)))  _Float16 v8h;
typedef __attribute__((ext_vector_type(4)))  _Float16 v4h;
typedef __attribute__((ext_vector_type(8)))  float    v8f;

__device__ __forceinline__ v16h cat16(v8h a, v8h b) {
  return __builtin_shufflevector(a, b, 0,1,2,3,4,5,6,7,8,9,10,11,12,13,14,15);
}

#define NTOK 8192
#define DDIM 1024
#define FDIM 4096
#define NEXP 4
#define HS_STRIDE 4104   // 4096 + 8 halves: keeps 16B alignment, spreads banks

// ---------------------------------------------------------------- convert x
__global__ void convert_x(const float* __restrict__ src, _Float16* __restrict__ dst, int n4) {
  int i = blockIdx.x * 256 + threadIdx.x;
  if (i < n4) {
    float4 v = ((const float4*)src)[i];
    v4h o = { (_Float16)v.x, (_Float16)v.y, (_Float16)v.z, (_Float16)v.w };
    ((v4h*)dst)[i] = o;
  }
}

// ------------------------------------------- fp32 [R][C] -> f16 [C][R] per e
__global__ void transpose_cvt(const float* __restrict__ src, _Float16* __restrict__ dst,
                              int R, int C) {
  __shared__ float tile[32][33];
  const size_t eoff = (size_t)blockIdx.z * R * C;
  const float* s = src + eoff;
  _Float16*    d = dst + eoff;
  const int c0 = blockIdx.x * 32, r0 = blockIdx.y * 32;
  const int tx = threadIdx.x, ty = threadIdx.y;
  #pragma unroll
  for (int i = 0; i < 32; i += 8)
    tile[ty + i][tx] = s[(size_t)(r0 + ty + i) * C + (c0 + tx)];
  __syncthreads();
  #pragma unroll
  for (int i = 0; i < 32; i += 8)
    d[(size_t)(c0 + ty + i) * R + (r0 + tx)] = (_Float16)tile[tx][ty + i];
}

// -------------------------------------------------- gating: top-2 binary mask
__global__ void gating_kernel(const float* __restrict__ x, const float* __restrict__ Wg,
                              const float* __restrict__ bg, float* __restrict__ mask) {
  const int wave = threadIdx.x >> 5, lane = threadIdx.x & 31;
  const int token = blockIdx.x * 8 + wave;
  const float* xr = x + (size_t)token * DDIM;
  float a0 = 0.f, a1 = 0.f, a2 = 0.f, a3 = 0.f;
  for (int dd = lane; dd < DDIM; dd += 32) {
    const float xv = xr[dd];
    const float4 wv = *(const float4*)(Wg + dd * 4);
    a0 += xv * wv.x; a1 += xv * wv.y; a2 += xv * wv.z; a3 += xv * wv.w;
  }
  #pragma unroll
  for (int off = 16; off > 0; off >>= 1) {
    a0 += __shfl_xor(a0, off, 32);
    a1 += __shfl_xor(a1, off, 32);
    a2 += __shfl_xor(a2, off, 32);
    a3 += __shfl_xor(a3, off, 32);
  }
  if (lane == 0) {
    float lg[4] = { a0 + bg[0], a1 + bg[1], a2 + bg[2], a3 + bg[3] };
    int i1 = 0;                                   // argmax, lowest index on tie
    for (int i = 1; i < 4; ++i) if (lg[i] > lg[i1]) i1 = i;
    int i2 = (i1 == 0) ? 1 : 0;                   // second best
    for (int i = 0; i < 4; ++i) if (i != i1 && lg[i] > lg[i2]) i2 = i;
    float m[4] = { 0.f, 0.f, 0.f, 0.f };
    m[i1] = 1.f; m[i2] = 1.f;
    *(float4*)(mask + (size_t)token * 4) = make_float4(m[0], m[1], m[2], m[3]);
  }
}

// ------------------------------------------------------- fused MoE dual GEMM
// One workgroup (8 waves) per 16-token tile. Per expert:
//   Phase 1: Hs = mask * swish(X @ W1[e] + b1[e])   (16 x 4096 f16 in LDS)
//   Phase 2: outAcc += Hs @ W2[e]                   (16 x 1024 f32 in VGPRs)
// Epilogue adds sum_e mask_e * b2[e]. X A-fragments come straight from global
// (32KB unique, WGP$-resident) so LDS stays small enough for 2 blocks/WGP.
__global__ void __launch_bounds__(256, 2)
moe_kernel(const _Float16* __restrict__ Xh, const _Float16* __restrict__ W1t,
           const _Float16* __restrict__ W2t, const float* __restrict__ b1,
           const float* __restrict__ b2, const float* __restrict__ maskG,
           float* __restrict__ out) {
  __shared__ _Float16 Hs[16 * HS_STRIDE];   // 131.3 KB
  __shared__ float    Ms[16 * 4];

  const int tid  = threadIdx.x;
  const int wave = tid >> 5;
  const int lane = tid & 31;
  const int col  = lane & 15;   // A row (token) / B+C column
  const int hi   = lane >> 4;   // K-half selector per WMMA f16 layout
  const size_t tokBase = (size_t)blockIdx.x * 16;

  // Per-lane A-fragment base for the X tile (row = col, K chunk by hi).
  const _Float16* xrow = Xh + (tokBase + col) * DDIM + hi * 8;

  if (tid < 64) Ms[tid] = maskG[tokBase * 4 + tid];
  __syncthreads();

  v8f outAcc[8];
  #pragma unroll
  for (int i = 0; i < 8; ++i) { v8f z = {}; outAcc[i] = z; }

  for (int e = 0; e < NEXP; ++e) {
    float mrow[8];                                // mask for C rows j + hi*8
    #pragma unroll
    for (int j = 0; j < 8; ++j) mrow[j] = Ms[(j + hi * 8) * 4 + e];

    // ---- Phase 1: wave owns 512 f-columns; 2 col-tiles share each A frag ----
    const _Float16* w1e = W1t + (size_t)e * FDIM * DDIM;
    const float*    b1e = b1  + (size_t)e * FDIM;
    for (int ftp = 0; ftp < 16; ++ftp) {
      const int f0 = wave * 512 + ftp * 32;
      const _Float16* bp0 = w1e + (size_t)(f0 + col) * DDIM + hi * 8;
      const _Float16* bp1 = bp0 + (size_t)16 * DDIM;
      v8f acc0 = {}, acc1 = {};
      #pragma unroll 2
      for (int k = 0; k < DDIM; k += 32) {
        const v16h aa = cat16(*(const v8h*)(xrow + k), *(const v8h*)(xrow + k + 16));
        const v16h b0 = cat16(*(const v8h*)(bp0 + k), *(const v8h*)(bp0 + k + 16));
        const v16h b1v = cat16(*(const v8h*)(bp1 + k), *(const v8h*)(bp1 + k + 16));
        acc0 = __builtin_amdgcn_wmma_f32_16x16x32_f16(
            false, aa, false, b0, (short)0, acc0, false, false);
        acc1 = __builtin_amdgcn_wmma_f32_16x16x32_f16(
            false, aa, false, b1v, (short)0, acc1, false, false);
      }
      const float biasA = b1e[f0 + col];
      const float biasB = b1e[f0 + 16 + col];
      #pragma unroll
      for (int j = 0; j < 8; ++j) {
        const int r = (j + hi * 8) * HS_STRIDE + f0 + col;
        float c0 = acc0[j] + biasA;
        float c1 = acc1[j] + biasB;
        // swish via v_exp + v_rcp (no IEEE divide), with binary mask folded in
        float h0 = c0 * __builtin_amdgcn_rcpf(1.f + __expf(-c0)) * mrow[j];
        float h1 = c1 * __builtin_amdgcn_rcpf(1.f + __expf(-c1)) * mrow[j];
        Hs[r]      = (_Float16)h0;
        Hs[r + 16] = (_Float16)h1;
      }
    }
    __syncthreads();

    // ---- Phase 2: wave owns 128 d-columns (8 tiles of 16), persistent acc ----
    const _Float16* w2e = W2t + (size_t)e * DDIM * FDIM;
    for (int k = 0; k < FDIM; k += 32) {
      const v8h a0 = *(const v8h*)(&Hs[col * HS_STRIDE + k + hi * 8]);
      const v8h a1 = *(const v8h*)(&Hs[col * HS_STRIDE + k + 16 + hi * 8]);
      const v16h aa = cat16(a0, a1);
      #pragma unroll
      for (int dt = 0; dt < 8; ++dt) {
        const int d = wave * 128 + dt * 16 + col;
        const _Float16* bp = w2e + (size_t)d * FDIM + k + hi * 8;
        const v16h bb = cat16(*(const v8h*)(bp), *(const v8h*)(bp + 16));
        outAcc[dt] = __builtin_amdgcn_wmma_f32_16x16x32_f16(
            false, aa, false, bb, (short)0, outAcc[dt], false, false);
      }
    }
    __syncthreads();
  }

  // Epilogue: out = outAcc + sum_e mask_e * b2[e][d]
  #pragma unroll
  for (int dt = 0; dt < 8; ++dt) {
    const int d = wave * 128 + dt * 16 + col;
    const float b20 = b2[0 * DDIM + d], b21 = b2[1 * DDIM + d];
    const float b22 = b2[2 * DDIM + d], b23 = b2[3 * DDIM + d];
    #pragma unroll
    for (int j = 0; j < 8; ++j) {
      const int row = j + hi * 8;
      const float bias = Ms[row * 4 + 0] * b20 + Ms[row * 4 + 1] * b21 +
                         Ms[row * 4 + 2] * b22 + Ms[row * 4 + 3] * b23;
      out[(tokBase + row) * DDIM + d] = outAcc[dt][j] + bias;
    }
  }
}

// ---------------------------------------------------------------------------
extern "C" void kernel_launch(void* const* d_in, const int* in_sizes, int n_in,
                              void* d_out, int out_size, void* d_ws, size_t ws_size,
                              hipStream_t stream) {
  const float* x  = (const float*)d_in[0];
  const float* Wg = (const float*)d_in[1];
  const float* bg = (const float*)d_in[2];
  const float* W1 = (const float*)d_in[3];
  const float* b1 = (const float*)d_in[4];
  const float* W2 = (const float*)d_in[5];
  const float* b2 = (const float*)d_in[6];
  float* out = (float*)d_out;

  char* ws = (char*)d_ws;
  _Float16* Xh   = (_Float16*)(ws);
  _Float16* W1t  = (_Float16*)(ws + (size_t)16 * 1024 * 1024);
  _Float16* W2t  = (_Float16*)(ws + (size_t)48 * 1024 * 1024);
  float*    mask = (float*)   (ws + (size_t)80 * 1024 * 1024);

  // One-time (per call) precision conversion + weight transposes.
  convert_x<<<(NTOK * DDIM / 4 + 255) / 256, 256, 0, stream>>>(x, Xh, NTOK * DDIM / 4);
  transpose_cvt<<<dim3(FDIM / 32, DDIM / 32, NEXP), dim3(32, 8), 0, stream>>>(W1, W1t, DDIM, FDIM);
  transpose_cvt<<<dim3(DDIM / 32, FDIM / 32, NEXP), dim3(32, 8), 0, stream>>>(W2, W2t, FDIM, DDIM);

  // Gating -> binary top-2 mask.
  gating_kernel<<<NTOK / 8, 256, 0, stream>>>(x, Wg, bg, mask);

  // Fused all-expert dual GEMM with mask folding.
  moe_kernel<<<NTOK / 16, 256, 0, stream>>>(Xh, W1t, W2t, b1, b2, mask, out);
}
